// CL_model_65257733095574
// MI455X (gfx1250) — matrix operand order repaired and checked by reading, not compiled, not run on hardware
//
#include <hip/hip_runtime.h>
#include <hip/hip_bf16.h>

#define Bn 32
#define Pn 2048
#define Kn 20
#define NEG_INF (-3.402823466e38f)

typedef __attribute__((ext_vector_type(16))) _Float16 v16h;
typedef __attribute__((ext_vector_type(8)))  _Float16 v8h;
typedef __attribute__((ext_vector_type(8)))  float    v8f;

// ---------- CDNA5 helpers ----------

static __device__ __forceinline__ v8f wmma_f16(v16h a, v16h b, v8f c) {
  // v_wmma_f32_16x16x32_f16  (probe-verified signature, 8 args)
  return __builtin_amdgcn_wmma_f32_16x16x32_f16(false, a, false, b, (short)0, c,
                                                false, false);
}

// Per-wave LDS ordering: own LDS writes visible before own LDS reads.
static __device__ __forceinline__ void wave_lds_fence() {
  __builtin_amdgcn_wave_barrier();
  asm volatile("s_wait_dscnt 0x0" ::: "memory");
  __builtin_amdgcn_wave_barrier();
}

// Build a v16h fragment from two contiguous, 16B-aligned 8-half runs in LDS.
// Each run lowers to a single ds_load_b128.
static __device__ __forceinline__ v16h ld2x8(const _Float16* p0, const _Float16* p1) {
  v8h lo = *(const v8h*)p0;
  v8h hi = *(const v8h*)p1;
  v16h r;
#pragma unroll
  for (int e = 0; e < 8; ++e) { r[e] = lo[e]; r[8 + e] = hi[e]; }
  return r;
}

// 16-bit A-fragment (16x32 tile of a row-major f16 matrix in LDS).
// Lanes 0-15 hold K=0..15, lanes 16-31 K=16..31 (matches B-fragment order).
static __device__ __forceinline__ v16h load_afrag(const _Float16* M, int ldm,
                                                  int row, int kbase, int lane) {
  const int hi = (lane >> 4) & 1;
  const int base = row * ldm + kbase + 16 * hi;
  return ld2x8(M + base, M + base + 8);
}

// 16-bit B-fragment from a TRANSPOSED (col-major: MT[col][k]) f16 matrix in LDS.
// Lane's 16 halves are K=kbase+16*hi .. +15 -> one contiguous 32B run.
static __device__ __forceinline__ v16h load_bfrag_t(const _Float16* MT, int ldk,
                                                    int kbase, int col, int lane) {
  const int hi = (lane >> 4) & 1;
  const int base = col * ldk + kbase + 16 * hi;
  return ld2x8(MT + base, MT + base + 8);
}

static __device__ __forceinline__ float vmax8(v8f v) {
  float m = v[0];
#pragma unroll
  for (int i = 1; i < 8; ++i) m = fmaxf(m, v[i]);
  return m;
}

static __device__ __forceinline__ void topk_insert(float* tv, int* ti, float v, int id) {
  if (v <= tv[Kn - 1]) return;
  int p = Kn - 1;
  while (p > 0 && tv[p - 1] < v) { tv[p] = tv[p - 1]; ti[p] = ti[p - 1]; --p; }
  tv[p] = v; ti[p] = id;
}

// ---------- Kernel 1: KNN on pos (dim=3), 16 queries per block ----------
__global__ __launch_bounds__(128) void knn3_kernel(const float* __restrict__ pos,
                                                   int* __restrict__ idx) {
  __shared__ float sq[16][3];
  __shared__ float sc[64][3];
  __shared__ float S[16][64];
  __shared__ float tv[16][Kn];
  __shared__ int   ti[16][Kn];

  const int t = threadIdx.x;
  const int b = blockIdx.x >> 7;         // P/16 = 128 blocks per batch
  const int qbase = (blockIdx.x & 127) * 16;
  const float* posb = pos + (size_t)b * Pn * 3;

  if (t < 48) sq[t / 3][t % 3] = posb[(qbase + t / 3) * 3 + (t % 3)];
  if (t < 16) {
#pragma unroll
    for (int j = 0; j < Kn; ++j) { tv[t][j] = NEG_INF; ti[t][j] = 0; }
  }
  __syncthreads();

  for (int cb = 0; cb < Pn; cb += 64) {
    for (int i = t; i < 192; i += 128) sc[i / 3][i % 3] = posb[(cb + i / 3) * 3 + (i % 3)];
    __syncthreads();
    const int r = t >> 3;
    const int j0 = (t & 7) * 8;
#pragma unroll
    for (int jj = 0; jj < 8; ++jj) {
      int j = j0 + jj;
      float dx = sq[r][0] - sc[j][0];
      float dy = sq[r][1] - sc[j][1];
      float dz = sq[r][2] - sc[j][2];
      S[r][j] = -(dx * dx + dy * dy + dz * dz);   // ranking-equivalent to reference
    }
    __syncthreads();
    if (t < 16)
      for (int j = 0; j < 64; ++j) topk_insert(tv[t], ti[t], S[t][j], cb + j);
    __syncthreads();
  }
  for (int i = t; i < 16 * Kn; i += 128)
    idx[((size_t)b * Pn + qbase + i / Kn) * Kn + (i % Kn)] = ti[i / Kn][i % Kn];
}

// ---------- Kernel 2: edge conv 1 -> x1 (B,P,64). One wave per point ----------
__global__ __launch_bounds__(256) void edge1_kernel(const float* __restrict__ pos,
                                                    const int* __restrict__ idx1,
                                                    const float* __restrict__ W1,
                                                    const float* __restrict__ b1,
                                                    const float* __restrict__ g1,
                                                    const float* __restrict__ bt1,
                                                    const float* __restrict__ W2,
                                                    const float* __restrict__ b2,
                                                    float* __restrict__ x1out) {
  __shared__ __align__(16) _Float16 sW1T[64][32];   // transposed, K-padded 6->32
  __shared__ __align__(16) _Float16 sW2T[64][64];   // transposed [col][k]
  __shared__ float sb1[64], sg1[64], sbt1[64], sb2[64];
  __shared__ __align__(16) _Float16 sh[8][32][64];  // per-wave hidden staging

  const int t = threadIdx.x;
  for (int i = t; i < 64 * 32; i += 256) {
    int c = i >> 5, k = i & 31;
    sW1T[c][k] = (k < 6) ? (_Float16)W1[k * 64 + c] : (_Float16)0.0f;
  }
  for (int i = t; i < 64 * 64; i += 256) {
    int c = i >> 6, k = i & 63;
    sW2T[c][k] = (_Float16)W2[k * 64 + c];
  }
  if (t < 64) { sb1[t] = b1[t]; sg1[t] = g1[t]; sbt1[t] = bt1[t]; sb2[t] = b2[t]; }
  __syncthreads();

  const int w = t >> 5, lane = t & 31, lr = lane & 15, hi = lane >> 4;
  const int p = blockIdx.x * 8 + w;
  const int b = p >> 11, pi = p & (Pn - 1);
  const float* posb = pos + (size_t)b * Pn * 3;

  const float xi0 = posb[pi * 3 + 0], xi1 = posb[pi * 3 + 1], xi2 = posb[pi * 3 + 2];

  // A fragments for msg tiles (rows = neighbors; K: [xi, xj-xi], padded to 32)
  v16h a0, a1;
#pragma unroll
  for (int e = 0; e < 16; ++e) { a0[e] = (_Float16)0.0f; a1[e] = (_Float16)0.0f; }
  if (hi == 0) {                              // only lanes 0-15 carry K=0..5
    int j0 = idx1[(size_t)p * Kn + lr];
    int j1 = (lr < 4) ? idx1[(size_t)p * Kn + 16 + lr] : pi;  // pad rows -> self
    float x0 = posb[j0 * 3 + 0], y0 = posb[j0 * 3 + 1], z0 = posb[j0 * 3 + 2];
    float x1n = posb[j1 * 3 + 0], y1n = posb[j1 * 3 + 1], z1n = posb[j1 * 3 + 2];
    a0[0] = (_Float16)xi0; a0[1] = (_Float16)xi1; a0[2] = (_Float16)xi2;
    a0[3] = (_Float16)(x0 - xi0); a0[4] = (_Float16)(y0 - xi1); a0[5] = (_Float16)(z0 - xi2);
    a1[0] = (_Float16)xi0; a1[1] = (_Float16)xi1; a1[2] = (_Float16)xi2;
    a1[3] = (_Float16)(x1n - xi0); a1[4] = (_Float16)(y1n - xi1); a1[5] = (_Float16)(z1n - xi2);
  }

  // GEMM1: (32x32) x (32x64) -> affine+ReLU -> stage f16 hidden
#pragma unroll
  for (int nt = 0; nt < 4; ++nt) {
    v8f c0 = {}, c1 = {};
    v16h bf = load_bfrag_t(&sW1T[0][0], 32, 0, nt * 16 + lr, lane);
    c0 = wmma_f16(a0, bf, c0);
    c1 = wmma_f16(a1, bf, c1);
    const int col = nt * 16 + lr;
#pragma unroll
    for (int r = 0; r < 8; ++r) {
      int M = r + 8 * hi;
      float v0 = fmaxf((c0[r] + sb1[col]) * sg1[col] + sbt1[col], 0.0f);
      float v1 = fmaxf((c1[r] + sb1[col]) * sg1[col] + sbt1[col], 0.0f);
      sh[w][M][col]      = (_Float16)v0;
      sh[w][16 + M][col] = (_Float16)v1;
    }
  }
  wave_lds_fence();

  // GEMM2: (32x64) x (64x64), then max over the 20 valid rows (+b2 after max)
#pragma unroll
  for (int nt = 0; nt < 4; ++nt) {
    v8f d0 = {}, d1 = {};
#pragma unroll
    for (int kt = 0; kt < 2; ++kt) {
      v16h fa0 = load_afrag(&sh[w][0][0], 64, lr,      kt * 32, lane);
      v16h fa1 = load_afrag(&sh[w][0][0], 64, 16 + lr, kt * 32, lane);
      v16h fb  = load_bfrag_t(&sW2T[0][0], 64, kt * 32, nt * 16 + lr, lane);
      d0 = wmma_f16(fa0, fb, d0);
      d1 = wmma_f16(fa1, fb, d1);
    }
    float t0 = vmax8(d0);
    float t0c = fmaxf(t0, __shfl_xor(t0, 16));                 // rows 0-15
    float t1 = fmaxf(fmaxf(d1[0], d1[1]), fmaxf(d1[2], d1[3])); // rows 16-19 (lanes<16)
    float t1v = (hi == 0) ? t1 : NEG_INF;
    float t1c = fmaxf(t1v, __shfl_xor(t1v, 16));
    float m = fmaxf(t0c, t1c);
    if (lane < 16) {
      int col = nt * 16 + lr;
      x1out[(size_t)p * 64 + col] = m + sb2[col];
    }
  }
}

// ---------- Kernel 3: KNN on x1 (dim=64) with WMMA Gram tiles ----------
__global__ __launch_bounds__(128) void knn64_kernel(const float* __restrict__ x1,
                                                    int* __restrict__ idx) {
  __shared__ __align__(16) _Float16 Qh[16][64];
  __shared__ __align__(16) _Float16 Ch[64][64];
  __shared__ float cn[64];
  __shared__ float S[16][64];
  __shared__ float tv[16][Kn];
  __shared__ int   ti[16][Kn];

  const int t = threadIdx.x;
  const int b = blockIdx.x >> 7;
  const int qbase = (blockIdx.x & 127) * 16;
  const float* x1b = x1 + (size_t)b * Pn * 64;

  for (int i = t; i < 16 * 64; i += 128)
    Qh[i >> 6][i & 63] = (_Float16)x1b[(size_t)(qbase + (i >> 6)) * 64 + (i & 63)];
  if (t < 16) {
#pragma unroll
    for (int j = 0; j < Kn; ++j) { tv[t][j] = NEG_INF; ti[t][j] = 0; }
  }
  __syncthreads();

  const int w = t >> 5, lane = t & 31, lr = lane & 15, hi = lane >> 4;

  for (int cb = 0; cb < Pn; cb += 64) {
    if (t == 0 && cb + 64 < Pn)
      __builtin_prefetch(x1b + (size_t)(cb + 64) * 64, 0, 1);  // global_prefetch
    for (int i = t; i < 64 * 64; i += 128)
      Ch[i >> 6][i & 63] = (_Float16)x1b[(size_t)cb * 64 + i]; // contiguous chunk
    __syncthreads();
    if (t < 64) {
      float s = 0.0f;
#pragma unroll
      for (int k = 0; k < 64; ++k) { float v = (float)Ch[t][k]; s += v * v; }
      cn[t] = s;
    }
    __syncthreads();

    // wave w computes the 16x16 Gram tile for candidate columns [w*16, w*16+16)
    v8f g = {};
#pragma unroll
    for (int kt = 0; kt < 2; ++kt) {
      v16h fa = load_afrag(&Qh[0][0], 64, lr, kt * 32, lane);
      // B[k][n] = Cand[n][k] -> row (w*16+lr) of Ch, contiguous along k
      const int base = (w * 16 + lr) * 64 + kt * 32 + 16 * hi;
      v16h fb = ld2x8(&Ch[0][0] + base, &Ch[0][0] + base + 8);
      g = wmma_f16(fa, fb, g);
    }
#pragma unroll
    for (int r = 0; r < 8; ++r) {
      int M = r + 8 * hi, col = w * 16 + lr;
      S[M][col] = 2.0f * g[r] - cn[col];   // |q|^2 constant per row -> dropped
    }
    __syncthreads();
    if (t < 16)
      for (int j = 0; j < 64; ++j) topk_insert(tv[t], ti[t], S[t][j], cb + j);
    __syncthreads();
  }
  for (int i = t; i < 16 * Kn; i += 128)
    idx[((size_t)b * Pn + qbase + i / Kn) * Kn + (i % Kn)] = ti[i / Kn][i % Kn];
}

// ---------- Kernel 4: edge conv 2 -> x2 (B,P,128). One wave per point ----------
__global__ __launch_bounds__(64) void edge2_kernel(const float* __restrict__ x1,
                                                   const int* __restrict__ idx2,
                                                   const float* __restrict__ W3,
                                                   const float* __restrict__ b3,
                                                   float* __restrict__ x2out) {
  __shared__ __align__(16) _Float16 sW3T[128][128];   // transposed [col][k]
  __shared__ float sb3[128];
  __shared__ __align__(16) _Float16 sm[2][32][128];

  const int t = threadIdx.x;
  for (int i = t; i < 128 * 128; i += 64) {
    int c = i >> 7, k = i & 127;
    sW3T[c][k] = (_Float16)W3[k * 128 + c];
  }
  for (int i = t; i < 128; i += 64) sb3[i] = b3[i];
  __syncthreads();

  const int w = t >> 5, lane = t & 31, lr = lane & 15, hi = lane >> 4;
  const int p = blockIdx.x * 2 + w;
  const int b = p >> 11, pi = p & (Pn - 1);
  const float* x1b = x1 + (size_t)b * Pn * 64;

  { // each lane stages one msg row: [xi(64) | xj-xi(64)]
    const int m = lane;
    const int j = (m < Kn) ? idx2[(size_t)p * Kn + m] : pi;
#pragma unroll 8
    for (int c = 0; c < 64; ++c) {
      float xiv = x1b[(size_t)pi * 64 + c];
      float xjv = x1b[(size_t)j * 64 + c];
      sm[w][m][c]      = (_Float16)xiv;
      sm[w][m][64 + c] = (_Float16)(xjv - xiv);
    }
  }
  wave_lds_fence();

#pragma unroll
  for (int nt = 0; nt < 8; ++nt) {
    v8f d0 = {}, d1 = {};
#pragma unroll
    for (int kt = 0; kt < 4; ++kt) {
      v16h fa0 = load_afrag(&sm[w][0][0], 128, lr,      kt * 32, lane);
      v16h fa1 = load_afrag(&sm[w][0][0], 128, 16 + lr, kt * 32, lane);
      v16h fb  = load_bfrag_t(&sW3T[0][0], 128, kt * 32, nt * 16 + lr, lane);
      d0 = wmma_f16(fa0, fb, d0);
      d1 = wmma_f16(fa1, fb, d1);
    }
    float t0 = vmax8(d0);
    float t0c = fmaxf(t0, __shfl_xor(t0, 16));
    float t1 = fmaxf(fmaxf(d1[0], d1[1]), fmaxf(d1[2], d1[3]));
    float t1v = (hi == 0) ? t1 : NEG_INF;
    float t1c = fmaxf(t1v, __shfl_xor(t1v, 16));
    float m = fmaxf(t0c, t1c);
    if (lane < 16) {
      int col = nt * 16 + lr;
      x2out[(size_t)p * 128 + col] = m + sb3[col];
    }
  }
}

// ---------- Kernel 5: [x1|x2] @ Wl, partial column-max over 128 points ----------
__global__ __launch_bounds__(128) void final_kernel(const float* __restrict__ x1,
                                                    const float* __restrict__ x2,
                                                    const float* __restrict__ Wl,
                                                    float* __restrict__ partial) {
  __shared__ __align__(16) _Float16 sA[8][16][192];
  __shared__ float smax[8][128];

  const int t = threadIdx.x;
  const int b = blockIdx.x >> 4;
  const int rbase = (blockIdx.x & 15) * 128;

  { // thread t stages point row rbase+t (each wave fills exactly its own tiles)
    const int row = t, pi = rbase + row, tl = row >> 4, rr = row & 15;
    const float* r1 = x1 + ((size_t)b * Pn + pi) * 64;
    const float* r2 = x2 + ((size_t)b * Pn + pi) * 128;
#pragma unroll 8
    for (int c = 0; c < 64; ++c)  sA[tl][rr][c] = (_Float16)r1[c];
#pragma unroll 8
    for (int c = 0; c < 128; ++c) sA[tl][rr][64 + c] = (_Float16)r2[c];
  }
  wave_lds_fence();

  const int w = t >> 5, lane = t & 31, lr = lane & 15, hi = lane >> 4;
#pragma unroll
  for (int nt = 0; nt < 8; ++nt) {
    v8f d0 = {}, d1 = {};
#pragma unroll
    for (int kt = 0; kt < 6; ++kt) {
      // Wl fragment (f32 -> f16 on the fly, hot in L2), reused by both row tiles
      v16h fb;
#pragma unroll
      for (int e = 0; e < 16; ++e) {
        int K = kt * 32 + 16 * hi + e;
        fb[e] = (_Float16)Wl[(size_t)K * 128 + nt * 16 + lr];
      }
      v16h fa0 = load_afrag(&sA[2 * w][0][0],     192, lr, kt * 32, lane);
      v16h fa1 = load_afrag(&sA[2 * w + 1][0][0], 192, lr, kt * 32, lane);
      d0 = wmma_f16(fa0, fb, d0);
      d1 = wmma_f16(fa1, fb, d1);
    }
    float t0 = vmax8(d0);
    float m0 = fmaxf(t0, __shfl_xor(t0, 16));     // max over all 16 rows
    float t1 = vmax8(d1);
    float m1 = fmaxf(t1, __shfl_xor(t1, 16));
    if (lane < 16) {
      smax[2 * w][nt * 16 + lr]     = m0;
      smax[2 * w + 1][nt * 16 + lr] = m1;
    }
  }
  __syncthreads();
  { // reduce 8 tiles -> block partial
    float v = NEG_INF;
#pragma unroll
    for (int tl = 0; tl < 8; ++tl) v = fmaxf(v, smax[tl][t]);
    partial[(size_t)blockIdx.x * 128 + t] = v;
  }
}

// ---------- Kernel 6: reduce partials per batch, add bl ----------
__global__ __launch_bounds__(128) void reduce_kernel(const float* __restrict__ partial,
                                                     const float* __restrict__ bl,
                                                     float* __restrict__ out) {
  const int b = blockIdx.x, c = threadIdx.x;
  float v = NEG_INF;
#pragma unroll
  for (int i = 0; i < 16; ++i) v = fmaxf(v, partial[(size_t)(b * 16 + i) * 128 + c]);
  out[(size_t)b * 128 + c] = v + bl[c];
}

// ---------- Host side ----------
extern "C" void kernel_launch(void* const* d_in, const int* in_sizes, int n_in,
                              void* d_out, int out_size, void* d_ws, size_t ws_size,
                              hipStream_t stream) {
  (void)in_sizes; (void)n_in; (void)out_size; (void)ws_size;
  const float* pos = (const float*)d_in[0];
  const float* W1  = (const float*)d_in[1];
  const float* b1  = (const float*)d_in[2];
  const float* g1  = (const float*)d_in[3];
  const float* bt1 = (const float*)d_in[4];
  const float* W2  = (const float*)d_in[5];
  const float* b2  = (const float*)d_in[6];
  const float* W3  = (const float*)d_in[7];
  const float* b3  = (const float*)d_in[8];
  const float* Wl  = (const float*)d_in[9];
  const float* bl  = (const float*)d_in[10];
  float* out = (float*)d_out;

  char* base = (char*)d_ws;
  const size_t IDX_BYTES = (size_t)Bn * Pn * Kn * sizeof(int);   //  5,242,880
  const size_t X1_BYTES  = (size_t)Bn * Pn * 64 * sizeof(float); // 16,777,216
  const size_t X2_BYTES  = (size_t)Bn * Pn * 128 * sizeof(float);// 33,554,432
  int*   idx1    = (int*)  (base);
  float* x1buf   = (float*)(base + IDX_BYTES);
  int*   idx2    = (int*)  (base + IDX_BYTES + X1_BYTES);
  float* x2buf   = (float*)(base + IDX_BYTES + X1_BYTES + IDX_BYTES);
  float* partial = (float*)(base + IDX_BYTES + X1_BYTES + IDX_BYTES + X2_BYTES);

  const int NPTS = Bn * Pn;                       // 65536
  knn3_kernel  <<<Bn * (Pn / 16), 128, 0, stream>>>(pos, idx1);
  edge1_kernel <<<NPTS / 8, 256, 0, stream>>>(pos, idx1, W1, b1, g1, bt1, W2, b2, x1buf);
  knn64_kernel <<<Bn * (Pn / 16), 128, 0, stream>>>(x1buf, idx2);
  edge2_kernel <<<NPTS / 2, 64, 0, stream>>>(x1buf, idx2, W3, b3, x2buf);
  final_kernel <<<Bn * 16, 128, 0, stream>>>(x1buf, x2buf, Wl, partial);
  reduce_kernel<<<Bn, 128, 0, stream>>>(partial, bl, out);
}
